// GnnMapping_45037027066129
// MI455X (gfx1250) — matrix operand order repaired and checked by reading, not compiled, not run on hardware
//
#include <hip/hip_runtime.h>
#include <hip/hip_bf16.h>

// ---------------- problem constants ----------------
#define N_NODES 50000
#define N_EDGES 300000
#define NB      256      // graphs
#define F1      128      // feat dim
#define F2      256      // gnn width
#define NL      10       // layers
#define NA      2016     // actions
#define NEG_SLOPE 0.01f
#define GN_EPS  1e-5f

typedef __attribute__((ext_vector_type(16))) __bf16 v16bf;
typedef __attribute__((ext_vector_type(8)))  __bf16 v8bf;
typedef __attribute__((ext_vector_type(8)))  float  v8f;

// ---------------- small utility kernels ----------------
__global__ void k_zero_f32(float* __restrict__ p, int n) {
    int i = blockIdx.x * 256 + threadIdx.x;
    if (i < n) p[i] = 0.0f;
}
__global__ void k_zero_u32(unsigned* __restrict__ p, int n) {
    int i = blockIdx.x * 256 + threadIdx.x;
    if (i < n) p[i] = 0u;
}
__global__ void k_tobf(const float* __restrict__ a, __hip_bfloat16* __restrict__ o, int n) {
    int i = blockIdx.x * 256 + threadIdx.x;
    if (i < n) o[i] = __float2bfloat16(a[i]);
}

// ---------------- weight packing into WMMA B-fragment layout ----------------
// Packed layout: for k-tile t (32 K values), column n, half h (lane>=16),
// 16 contiguous bf16 = B[t*32 + h*16 + j][n], j=0..15  -> one 32B load/lane.
__device__ __forceinline__ size_t pack_idx(int k, int n, int Ncols) {
    int t = k >> 5, w = k & 31, half = w >> 4, j = w & 15;
    return ((((size_t)t * Ncols) + n) * 2 + half) * 16 + j;
}

__global__ void k_pack(const float* __restrict__ B, __hip_bfloat16* __restrict__ Bp,
                       int K, int Ncols) {
    int i = blockIdx.x * 256 + threadIdx.x;
    if (i >= K * Ncols) return;
    int k = i / Ncols, n = i - k * Ncols;
    Bp[pack_idx(k, n, Ncols)] = __float2bfloat16(B[(size_t)k * Ncols + n]);
}

// Stack [Wl ; Wr] per layer into K=512 packed weights.
__global__ void k_pack_sage(const float* __restrict__ Wl, const float* __restrict__ Wr,
                            __hip_bfloat16* __restrict__ Bp) {
    int i = blockIdx.x * 256 + threadIdx.x;
    if (i >= NL * 512 * F2) return;
    int layer = i / (512 * F2);
    int rem   = i - layer * 512 * F2;
    int k = rem / F2, n = rem - k * F2;
    float v = (k < F2) ? Wl[(size_t)layer * F2 * F2 + (size_t)k * F2 + n]
                       : Wr[(size_t)layer * F2 * F2 + (size_t)(k - F2) * F2 + n];
    Bp[(size_t)layer * 512 * F2 + pack_idx(k, n, F2)] = __float2bfloat16(v);
}

// ---------------- embedding ----------------
__global__ void k_embed(const int* __restrict__ x, const float* __restrict__ emb,
                        float* __restrict__ h, __hip_bfloat16* __restrict__ hb) {
    int i = blockIdx.x * 256 + threadIdx.x;
    if (i >= N_NODES * F2) return;
    int n = i / F2, f = i - n * F2;
    int which = (f >= F1) ? 1 : 0;
    int ff    = (f >= F1) ? f - F1 : f;
    int tok   = x[n * 2 + which];
    float v   = (tok == 0) ? 0.0f : emb[(size_t)tok * F1 + ff];
    h[i]  = v;
    hb[i] = __float2bfloat16(v);
}

// ---------------- graph structure ----------------
__global__ void k_deg(const int* __restrict__ ei, float* __restrict__ deg) {
    int e = blockIdx.x * 256 + threadIdx.x;
    if (e >= N_EDGES) return;
    atomicAdd(&deg[ei[N_EDGES + e]], 1.0f);
}

__global__ void k_scatter(const int* __restrict__ ei, const float* __restrict__ h,
                          float* __restrict__ agg) {
    int i = blockIdx.x * 256 + threadIdx.x;               // E * (F2/4)
    if (i >= N_EDGES * (F2 / 4)) return;
    int e = i / (F2 / 4), c = (i - e * (F2 / 4)) * 4;
    int s = ei[e], d = ei[N_EDGES + e];
    const float4 hv = *reinterpret_cast<const float4*>(&h[(size_t)s * F2 + c]);
    float* dst = &agg[(size_t)d * F2 + c];
    atomicAdd(dst + 0, hv.x); atomicAdd(dst + 1, hv.y);
    atomicAdd(dst + 2, hv.z); atomicAdd(dst + 3, hv.w);
}

__global__ void k_aggb(const float* __restrict__ agg, const float* __restrict__ deg,
                       __hip_bfloat16* __restrict__ aggb) {
    int i = blockIdx.x * 256 + threadIdx.x;
    if (i >= N_NODES * F2) return;
    int n = i / F2;
    float dg = deg[n]; dg = dg < 1.0f ? 1.0f : dg;
    aggb[i] = __float2bfloat16(agg[i] / dg);
}

// ---------------- WMMA GEMM: out = act(A @ Bpack + bias) ----------------
// A = [A0 (K0 cols) | A1 (K-K0 cols)] bf16 row-major; M rows (mult of 16).
// Each block: 16 rows x 256 cols (8 waves x 2 subtiles). ACT: 0 none,
// 1 leaky, 2 leaky+residual (SAGE). K0/K/NCOLS compile-time -> the K loop
// fully unrolls: all LDS / global fragment addresses become immediate
// offsets off hoisted bases, and loads pipeline across the WMMAs.
template<int K0, int K, int NCOLS, int ACT>
__global__ __launch_bounds__(256) void k_gemm(
    const __hip_bfloat16* __restrict__ A0,
    const __hip_bfloat16* __restrict__ A1,
    const __hip_bfloat16* __restrict__ Bp,
    const float* __restrict__ bias,
    float* __restrict__ outF, __hip_bfloat16* __restrict__ outB)
{
    __shared__ alignas(32) __hip_bfloat16 As[16 * K];
    const int tid  = threadIdx.x;
    const int row0 = blockIdx.x * 16;
    constexpr int Kc = K >> 3;                    // 8-half chunks per row
    #pragma unroll
    for (int c = tid; c < 16 * Kc; c += 256) {    // stage A tile (b128 copies)
        int r  = c / Kc;
        int kc = (c - r * Kc) * 8;
        const __hip_bfloat16* src = (kc < K0)
            ? A0 + (size_t)(row0 + r) * K0 + kc
            : A1 + (size_t)(row0 + r) * (K - K0) + (kc - K0);
        *reinterpret_cast<uint4*>(&As[r * K + kc]) =
            *reinterpret_cast<const uint4*>(src);
    }
    __syncthreads();

    const int wave  = tid >> 5;
    const int lane  = tid & 31;
    const int ncol0 = blockIdx.y * 256 + wave * 32;
    if (ncol0 >= NCOLS) return;

    const int ml   = lane & 15;
    const int hi16 = lane >> 4;

    // hoisted bases: everything below uses compile-time immediate offsets
    const __hip_bfloat16* ap = &As[ml * K + hi16 * 16];
    const __hip_bfloat16* bp = Bp + (((size_t)(ncol0 + ml)) * 2 + hi16) * 16;

    v8f acc0, acc1;
    {
        float b0 = bias[ncol0 + ml];
        float b1 = bias[ncol0 + 16 + ml];
        #pragma unroll
        for (int j = 0; j < 8; ++j) { acc0[j] = b0; acc1[j] = b1; }
    }

    constexpr int KT = K >> 5;
    #pragma unroll
    for (int kt = 0; kt < KT; ++kt) {
        // A fragment: contiguous k = 16*half + s (matches packed B): two b128 LDS loads
        v8bf alo = *reinterpret_cast<const v8bf*>(ap + kt * 32);
        v8bf ahi = *reinterpret_cast<const v8bf*>(ap + kt * 32 + 8);
        v16bf a  = __builtin_shufflevector(alo, ahi,
                     0,1,2,3,4,5,6,7,8,9,10,11,12,13,14,15);
        // B fragments: subtile 1 base is a constant +512 elements (1024 B)
        v16bf b0 = *reinterpret_cast<const v16bf*>(bp + (size_t)kt * NCOLS * 32);
        v16bf b1 = *reinterpret_cast<const v16bf*>(bp + (size_t)kt * NCOLS * 32 + 512);
        acc0 = __builtin_amdgcn_wmma_f32_16x16x32_bf16(
                   false, a, false, b0, (short)0, acc0, false, false);
        acc1 = __builtin_amdgcn_wmma_f32_16x16x32_bf16(
                   false, a, false, b1, (short)0, acc1, false, false);
    }

    #pragma unroll
    for (int j = 0; j < 8; ++j) {
        int gr = row0 + j + hi16 * 8;             // C/D layout: row = vgpr + 8*(lane>=16)
        float v0 = acc0[j], v1 = acc1[j];
        if (ACT == 1) {
            v0 = v0 > 0.0f ? v0 : NEG_SLOPE * v0;
            v1 = v1 > 0.0f ? v1 : NEG_SLOPE * v1;
        } else if (ACT == 2) {                    // leaky(x)+x
            v0 = v0 > 0.0f ? 2.0f * v0 : (1.0f + NEG_SLOPE) * v0;
            v1 = v1 > 0.0f ? 2.0f * v1 : (1.0f + NEG_SLOPE) * v1;
        }
        size_t o0 = (size_t)gr * NCOLS + ncol0 + ml;
        outF[o0]      = v0;
        outF[o0 + 16] = v1;
        if (outB) {
            outB[o0]      = __float2bfloat16(v0);
            outB[o0 + 16] = __float2bfloat16(v1);
        }
    }
}

// ---------------- GraphNorm ----------------
__global__ void k_cnt(const int* __restrict__ batch, float* __restrict__ cnt) {
    int n = blockIdx.x * 256 + threadIdx.x;
    if (n >= N_NODES) return;
    atomicAdd(&cnt[batch[n]], 1.0f);
}
__global__ void k_gsum(const int* __restrict__ batch, const float* __restrict__ h,
                       float* __restrict__ sum) {
    int i = blockIdx.x * 256 + threadIdx.x;
    if (i >= N_NODES * (F2 / 4)) return;
    int n = i / (F2 / 4), c = (i - n * (F2 / 4)) * 4;
    int b = batch[n];
    const float4 hv = *reinterpret_cast<const float4*>(&h[(size_t)n * F2 + c]);
    float* dst = &sum[(size_t)b * F2 + c];
    atomicAdd(dst + 0, hv.x); atomicAdd(dst + 1, hv.y);
    atomicAdd(dst + 2, hv.z); atomicAdd(dst + 3, hv.w);
}
__global__ void k_mean(float* __restrict__ sum, const float* __restrict__ cnt) {
    int i = blockIdx.x * 256 + threadIdx.x;
    if (i >= NB * F2) return;
    float c = cnt[i / F2]; c = c < 1.0f ? 1.0f : c;
    sum[i] /= c;
}
__global__ void k_ovar(const int* __restrict__ batch, const float* __restrict__ h,
                       const float* __restrict__ mean, const float* __restrict__ scale,
                       float* __restrict__ o, float* __restrict__ var) {
    int i = blockIdx.x * 256 + threadIdx.x;
    if (i >= N_NODES * F2) return;
    int n = i / F2, f = i - n * F2;
    int b = batch[n];
    float ov = h[i] - mean[(size_t)b * F2 + f] * scale[f];
    o[i] = ov;
    atomicAdd(&var[(size_t)b * F2 + f], ov * ov);
}
__global__ void k_gnfin(const int* __restrict__ batch, const float* __restrict__ o,
                        const float* __restrict__ var, const float* __restrict__ cnt,
                        const float* __restrict__ w, const float* __restrict__ bi,
                        float* __restrict__ h) {
    int i = blockIdx.x * 256 + threadIdx.x;
    if (i >= N_NODES * F2) return;
    int n = i / F2, f = i - n * F2;
    int b = batch[n];
    float c = cnt[b]; c = c < 1.0f ? 1.0f : c;
    float vv = var[(size_t)b * F2 + f] / c + GN_EPS;
    h[i] = w[f] * o[i] * rsqrtf(vv) + bi[f];
}

// ---------------- segment softmax aggregation ----------------
__device__ __forceinline__ unsigned encf(float x) {
    int i = __float_as_int(x);
    return (i >= 0) ? ((unsigned)i | 0x80000000u) : ~(unsigned)i;
}
__device__ __forceinline__ float decf(unsigned u) {
    int i = (u & 0x80000000u) ? (int)(u & 0x7fffffffu) : (int)~u;
    return __int_as_float(i);
}
__global__ void k_smax(const int* __restrict__ batch, const float* __restrict__ h,
                       const float* __restrict__ t, unsigned* __restrict__ mEnc) {
    int i = blockIdx.x * 256 + threadIdx.x;
    if (i >= N_NODES * F2) return;
    int n = i / F2, f = i - n * F2;
    atomicMax(&mEnc[(size_t)batch[n] * F2 + f], encf(h[i] * t[0]));
}
__global__ void k_sexp(const int* __restrict__ batch, const float* __restrict__ h,
                       const float* __restrict__ t, const unsigned* __restrict__ mEnc,
                       float* __restrict__ ex, float* __restrict__ den) {
    int i = blockIdx.x * 256 + threadIdx.x;
    if (i >= N_NODES * F2) return;
    int n = i / F2, f = i - n * F2;
    int b = batch[n];
    float e = expf(h[i] * t[0] - decf(mEnc[(size_t)b * F2 + f]));
    ex[i] = e;
    atomicAdd(&den[(size_t)b * F2 + f], e);
}
__global__ void k_sagg(const int* __restrict__ batch, const float* __restrict__ h,
                       const float* __restrict__ ex, const float* __restrict__ den,
                       float* __restrict__ g) {
    int i = blockIdx.x * 256 + threadIdx.x;
    if (i >= N_NODES * F2) return;
    int n = i / F2, f = i - n * F2;
    int b = batch[n];
    atomicAdd(&g[(size_t)b * F2 + f], ex[i] / den[(size_t)b * F2 + f] * h[i]);
}

// ---------------- driver ----------------
static inline int cdiv(long long a, int b) { return (int)((a + b - 1) / b); }

extern "C" void kernel_launch(void* const* d_in, const int* in_sizes, int n_in,
                              void* d_out, int out_size, void* d_ws, size_t ws_size,
                              hipStream_t stream) {
    (void)in_sizes; (void)n_in; (void)out_size; (void)ws_size;
    const int*   x    = (const int*)  d_in[0];
    const int*   ei   = (const int*)  d_in[1];
    const int*   bat  = (const int*)  d_in[2];
    const float* emb  = (const float*)d_in[3];
    const float* Wl   = (const float*)d_in[4];
    const float* bl   = (const float*)d_in[5];
    const float* Wr   = (const float*)d_in[6];
    const float* gnw  = (const float*)d_in[7];
    const float* gnb  = (const float*)d_in[8];
    const float* gns  = (const float*)d_in[9];
    const float* tpt  = (const float*)d_in[10];
    const float* W1   = (const float*)d_in[11];
    const float* b1   = (const float*)d_in[12];
    const float* W2   = (const float*)d_in[13];
    const float* b2   = (const float*)d_in[14];
    const float* W3   = (const float*)d_in[15];
    const float* b3   = (const float*)d_in[16];
    float* out = (float*)d_out;

    // workspace carve (256B aligned)
    char* base = (char*)d_ws;
    size_t off = 0;
    auto carve = [&](size_t bytes) -> void* {
        void* p = base + off;
        off = (off + bytes + 255) & ~(size_t)255;
        return p;
    };
    float*          h     = (float*)carve((size_t)N_NODES * F2 * 4);
    __hip_bfloat16* hb    = (__hip_bfloat16*)carve((size_t)N_NODES * F2 * 2);
    float*          agg   = (float*)carve((size_t)N_NODES * F2 * 4);   // also o / ex
    __hip_bfloat16* aggb  = (__hip_bfloat16*)carve((size_t)N_NODES * F2 * 2);
    float*          deg   = (float*)carve((size_t)N_NODES * 4);
    __hip_bfloat16* WsP   = (__hip_bfloat16*)carve((size_t)NL * 512 * F2 * 2);
    __hip_bfloat16* W1p   = (__hip_bfloat16*)carve((size_t)F2 * F2 * 2);
    __hip_bfloat16* W2p   = (__hip_bfloat16*)carve((size_t)F2 * F1 * 2);
    __hip_bfloat16* W3p   = (__hip_bfloat16*)carve((size_t)F1 * NA * 2);
    float*          cnt   = (float*)carve((size_t)NB * 4);
    float*          sum   = (float*)carve((size_t)NB * F2 * 4);        // -> mean
    float*          var   = (float*)carve((size_t)NB * F2 * 4);
    unsigned*       mEnc  = (unsigned*)carve((size_t)NB * F2 * 4);
    float*          den   = (float*)carve((size_t)NB * F2 * 4);
    float*          g     = (float*)carve((size_t)NB * F2 * 4);
    __hip_bfloat16* gb    = (__hip_bfloat16*)carve((size_t)NB * F2 * 2);
    float*          y1    = (float*)carve((size_t)NB * F2 * 4);
    __hip_bfloat16* y1b   = (__hip_bfloat16*)carve((size_t)NB * F2 * 2);
    float*          y2    = (float*)carve((size_t)NB * F1 * 4);
    __hip_bfloat16* y2b   = (__hip_bfloat16*)carve((size_t)NB * F1 * 2);

    const int NF = N_NODES * F2;

    // weight packing (L2-resident, WMMA fragment layout)
    k_pack_sage<<<cdiv((long long)NL * 512 * F2, 256), 256, 0, stream>>>(Wl, Wr, WsP);
    k_pack<<<cdiv((long long)F2 * F2, 256), 256, 0, stream>>>(W1, W1p, F2, F2);
    k_pack<<<cdiv((long long)F2 * F1, 256), 256, 0, stream>>>(W2, W2p, F2, F1);
    k_pack<<<cdiv((long long)F1 * NA, 256), 256, 0, stream>>>(W3, W3p, F1, NA);

    // node embedding + degrees
    k_embed<<<cdiv(NF, 256), 256, 0, stream>>>(x, emb, h, hb);
    k_zero_f32<<<cdiv(N_NODES, 256), 256, 0, stream>>>(deg, N_NODES);
    k_deg<<<cdiv(N_EDGES, 256), 256, 0, stream>>>(ei, deg);

    // 10x SAGEConv: scatter-mean then fused K=512 WMMA GEMM (in-place h/hb)
    for (int l = 0; l < NL; ++l) {
        k_zero_f32<<<cdiv(NF, 256), 256, 0, stream>>>(agg, NF);
        k_scatter<<<cdiv((long long)N_EDGES * (F2 / 4), 256), 256, 0, stream>>>(ei, h, agg);
        k_aggb<<<cdiv(NF, 256), 256, 0, stream>>>(agg, deg, aggb);
        k_gemm<256, 512, F2, 2><<<dim3(N_NODES / 16, 1), 256, 0, stream>>>(
            aggb, hb, WsP + (size_t)l * 512 * F2, bl + (size_t)l * F2, h, hb);
    }

    // GraphNorm
    k_zero_f32<<<cdiv(NB, 256), 256, 0, stream>>>(cnt, NB);
    k_zero_f32<<<cdiv(NB * F2, 256), 256, 0, stream>>>(sum, NB * F2);
    k_cnt<<<cdiv(N_NODES, 256), 256, 0, stream>>>(bat, cnt);
    k_gsum<<<cdiv((long long)N_NODES * (F2 / 4), 256), 256, 0, stream>>>(bat, h, sum);
    k_mean<<<cdiv(NB * F2, 256), 256, 0, stream>>>(sum, cnt);
    k_zero_f32<<<cdiv(NB * F2, 256), 256, 0, stream>>>(var, NB * F2);
    k_ovar<<<cdiv(NF, 256), 256, 0, stream>>>(bat, h, sum, gns, agg, var);
    k_gnfin<<<cdiv(NF, 256), 256, 0, stream>>>(bat, agg, var, cnt, gnw, gnb, h);

    // segment softmax aggregation
    k_zero_u32<<<cdiv(NB * F2, 256), 256, 0, stream>>>(mEnc, NB * F2);
    k_smax<<<cdiv(NF, 256), 256, 0, stream>>>(bat, h, tpt, mEnc);
    k_zero_f32<<<cdiv(NB * F2, 256), 256, 0, stream>>>(den, NB * F2);
    k_sexp<<<cdiv(NF, 256), 256, 0, stream>>>(bat, h, tpt, mEnc, agg, den);
    k_zero_f32<<<cdiv(NB * F2, 256), 256, 0, stream>>>(g, NB * F2);
    k_sagg<<<cdiv(NF, 256), 256, 0, stream>>>(bat, h, agg, den, g);
    k_tobf<<<cdiv(NB * F2, 256), 256, 0, stream>>>(g, gb, NB * F2);

    // MLP head (WMMA)
    k_gemm<F2, F2, F2, 1><<<dim3(NB / 16, 1), 256, 0, stream>>>(
        gb, nullptr, W1p, b1, y1, y1b);
    k_gemm<F2, F2, F1, 1><<<dim3(NB / 16, 1), 256, 0, stream>>>(
        y1b, nullptr, W2p, b2, y2, y2b);
    k_gemm<F1, F1, NA, 0><<<dim3(NB / 16, cdiv(NA, 256)), 256, 0, stream>>>(
        y2b, nullptr, W3p, b3, out, nullptr);
}
